// RPEAttention_28552942584353
// MI455X (gfx1250) — compile-verified, matrix operand order checked by reading, not run on hardware
//
#include <hip/hip_runtime.h>
#include <hip/hip_fp16.h>

typedef __attribute__((ext_vector_type(16))) _Float16 v16h;
typedef __attribute__((ext_vector_type(8)))  _Float16 v8h;
typedef __attribute__((ext_vector_type(8)))  float    v8f;

#define B_   2
#define D_   4
#define T_   192
#define C_   512
#define H_   8
#define TE_  512
#define F_   64
#define NG_  32
#define SCALE_ 0.125f
#define EPS_ 1e-5f

// ---------------------------------------------------------------------------
// Small conversion kernels
// ---------------------------------------------------------------------------
__global__ void k_cvt_t(const float* __restrict__ src, _Float16* __restrict__ dst,
                        int K, int N) {
  int i = blockIdx.x * 256 + threadIdx.x;
  if (i < K * N) {
    int k = i / N, n = i % N;
    dst[(long long)n * K + k] = (_Float16)src[i];
  }
}

__global__ void k_cvt(const float* __restrict__ src, _Float16* __restrict__ dst, int n) {
  int i = blockIdx.x * 256 + threadIdx.x;
  if (i < n) dst[i] = (_Float16)src[i];
}

// ---------------------------------------------------------------------------
// GroupNorm (32 groups over (C/32, T)) + transpose to (B*D, T, C), f32 + f16
// ---------------------------------------------------------------------------
__global__ void __launch_bounds__(256) k_groupnorm(
    const float* __restrict__ x, const float* __restrict__ w,
    const float* __restrict__ b, float* __restrict__ xt,
    _Float16* __restrict__ xth) {
  __shared__ float s1[256], s2[256];
  const int n = blockIdx.x / NG_, g = blockIdx.x % NG_;
  const int c0 = g * (C_ / NG_);
  const int tid = threadIdx.x;
  const int cnt = (C_ / NG_) * T_;
  float sum = 0.f, sq = 0.f;
  for (int i = tid; i < cnt; i += 256) {
    int c = c0 + i / T_, t = i % T_;
    float v = x[((long long)n * C_ + c) * T_ + t];
    sum += v; sq += v * v;
  }
  s1[tid] = sum; s2[tid] = sq; __syncthreads();
  for (int o = 128; o > 0; o >>= 1) {
    if (tid < o) { s1[tid] += s1[tid + o]; s2[tid] += s2[tid + o]; }
    __syncthreads();
  }
  float mu = s1[0] / (float)cnt;
  float var = s2[0] / (float)cnt - mu * mu;
  float rstd = rsqrtf(var + EPS_);
  for (int i = tid; i < cnt; i += 256) {
    int c = c0 + i / T_, t = i % T_;
    float v = x[((long long)n * C_ + c) * T_ + t];
    float y = (v - mu) * rstd * w[c] + b[c];
    long long o = ((long long)n * T_ + t) * C_ + c;
    xt[o] = y; xth[o] = (_Float16)y;
  }
}

// ---------------------------------------------------------------------------
// Generic batched WMMA GEMM: C = A(MxK,f16) * Bt(NxK,f16)^T  (+bias)
// 256 threads = 8 waves (2x4). Wave tile (BM/2)x(BN/4) of 16x16 WMMA tiles.
// EPI bits: 1 = add bias, 2 = write f16 to Ch, 4 = write f32 to C
// ---------------------------------------------------------------------------
template <int BM, int BN, int EPI>
__global__ void __launch_bounds__(256) k_gemm(
    const _Float16* __restrict__ A, const _Float16* __restrict__ Bt,
    float* __restrict__ C, _Float16* __restrict__ Ch,
    const float* __restrict__ bias,
    int M, int N, int K, long long sA, long long sB, long long sC) {
  constexpr int WM = BM / 32;  // m-frags per wave
  constexpr int WN = BN / 64;  // n-frags per wave
  __shared__ _Float16 As[BM * 32];
  __shared__ _Float16 Bs[BN * 32];
  const int tid = threadIdx.x;
  const int lane = tid & 31, wid = tid >> 5;
  const int wm = wid >> 2, wn = wid & 3;
  const int z = blockIdx.z;
  A += (long long)z * sA;
  Bt += (long long)z * sB;
  const long long cb = (long long)z * sC;
  const int row0 = blockIdx.y * BM, col0 = blockIdx.x * BN;
  const int lrow = lane & 15;
  const int koff = (lane < 16) ? 0 : 8;

  v8f acc[WM][WN] = {};

  for (int k0 = 0; k0 < K; k0 += 32) {
    for (int idx = tid; idx < BM * 4; idx += 256) {
      int r = idx >> 2, cc = (idx & 3) * 8;
      int g = row0 + r;
      v8h v = {};
      if (g < M) v = *(const v8h*)(A + (long long)g * K + k0 + cc);
      *(v8h*)(As + r * 32 + cc) = v;
    }
    for (int idx = tid; idx < BN * 4; idx += 256) {
      int r = idx >> 2, cc = (idx & 3) * 8;
      int g = col0 + r;
      v8h v = {};
      if (g < N) v = *(const v8h*)(Bt + (long long)g * K + k0 + cc);
      *(v8h*)(Bs + r * 32 + cc) = v;
    }
    __syncthreads();
    v16h af[WM], bf[WN];
#pragma unroll
    for (int i = 0; i < WM; ++i) {
      int r = wm * (BM / 2) + i * 16 + lrow;
      v8h lo = *(const v8h*)(As + r * 32 + koff);
      v8h hi = *(const v8h*)(As + r * 32 + koff + 16);
#pragma unroll
      for (int j = 0; j < 8; ++j) { af[i][j] = lo[j]; af[i][j + 8] = hi[j]; }
    }
#pragma unroll
    for (int i = 0; i < WN; ++i) {
      int r = wn * (BN / 4) + i * 16 + lrow;
      v8h lo = *(const v8h*)(Bs + r * 32 + koff);
      v8h hi = *(const v8h*)(Bs + r * 32 + koff + 16);
#pragma unroll
      for (int j = 0; j < 8; ++j) { bf[i][j] = lo[j]; bf[i][j + 8] = hi[j]; }
    }
#pragma unroll
    for (int i = 0; i < WM; ++i)
#pragma unroll
      for (int j = 0; j < WN; ++j)
        acc[i][j] = __builtin_amdgcn_wmma_f32_16x16x32_f16(
            false, af[i], false, bf[j], (short)0, acc[i][j], false, false);
    __syncthreads();
  }

#pragma unroll
  for (int i = 0; i < WM; ++i)
#pragma unroll
    for (int j = 0; j < WN; ++j) {
      int col = col0 + wn * (BN / 4) + j * 16 + lrow;
      int rb = row0 + wm * (BM / 2) + i * 16 + ((lane < 16) ? 0 : 8);
      if (col >= N) continue;
      float bv = (EPI & 1) ? bias[col] : 0.f;
#pragma unroll
      for (int r = 0; r < 8; ++r) {
        int row = rb + r;
        if (row < M) {
          float v = acc[i][j][r] + bv;
          if (EPI & 4) C[cb + (long long)row * N + col] = v;
          if (EPI & 2) Ch[cb + (long long)row * N + col] = (_Float16)v;
        }
      }
    }
}

// ---------------------------------------------------------------------------
// RPE GEMM: A[(b,t,s), c] = silu(temb[b,t,c] + de(b,t,s).dist_w[:,c] + dist_b[c])
// generated on the fly during LDS staging; R = A @ out_w + out_b  (f16 out).
// M = B*T*T = 73728, N = K = 512. BM = BN = 128.
// ---------------------------------------------------------------------------
__global__ void __launch_bounds__(256) k_rpe_gemm(
    const float* __restrict__ temb,   // (B*T, C)
    const float* __restrict__ dw,     // (3, C)
    const float* __restrict__ db,     // (C)
    const int* __restrict__ fdi,      // (B, T)
    const _Float16* __restrict__ Wt,  // (C, C) = out_w^T
    const float* __restrict__ ob,     // (C)
    _Float16* __restrict__ R) {       // (B*T*T, C)
  constexpr int BM = 128, BN = 128, WM = 4, WN = 2;
  __shared__ _Float16 As[BM * 32];
  __shared__ _Float16 Bs[BN * 32];
  __shared__ float sd0[BM], sd1[BM], sd2[BM];
  __shared__ int srow[BM];
  const int tid = threadIdx.x;
  const int lane = tid & 31, wid = tid >> 5;
  const int wm = wid >> 2, wn = wid & 3;
  const int row0 = blockIdx.y * BM, col0 = blockIdx.x * BN;
  const int lrow = lane & 15;
  const int koff = (lane < 16) ? 0 : 8;

  if (tid < BM) {
    int g = row0 + tid;
    int b = g / (T_ * T_);
    int rem = g % (T_ * T_);
    int t = rem / T_, s = rem % T_;
    int pdi = fdi[b * T_ + t] - fdi[b * T_ + s];
    float pdf = (float)pdi;
    sd0[tid] = log1pf(fmaxf(pdf, 0.f));
    sd1[tid] = log1pf(fmaxf(-pdf, 0.f));
    sd2[tid] = (pdi == 0) ? 1.f : 0.f;
    srow[tid] = b * T_ + t;
  }
  __syncthreads();

  v8f acc[WM][WN] = {};

  for (int k0 = 0; k0 < C_; k0 += 32) {
    for (int idx = tid; idx < BM * 4; idx += 256) {
      int r = idx >> 2, cc = (idx & 3) * 8;
      const float* te = temb + (long long)srow[r] * C_;
      float d0 = sd0[r], d1 = sd1[r], d2 = sd2[r];
      v8h v;
#pragma unroll
      for (int j = 0; j < 8; ++j) {
        int c = k0 + cc + j;
        float e = te[c] + d0 * dw[c] + d1 * dw[C_ + c] + d2 * dw[2 * C_ + c] + db[c];
        v[j] = (_Float16)(e / (1.f + __expf(-e)));
      }
      *(v8h*)(As + r * 32 + cc) = v;
    }
    for (int idx = tid; idx < BN * 4; idx += 256) {
      int r = idx >> 2, cc = (idx & 3) * 8;
      *(v8h*)(Bs + r * 32 + cc) =
          *(const v8h*)(Wt + (long long)(col0 + r) * C_ + k0 + cc);
    }
    __syncthreads();
    v16h af[WM], bf[WN];
#pragma unroll
    for (int i = 0; i < WM; ++i) {
      int r = wm * (BM / 2) + i * 16 + lrow;
      v8h lo = *(const v8h*)(As + r * 32 + koff);
      v8h hi = *(const v8h*)(As + r * 32 + koff + 16);
#pragma unroll
      for (int j = 0; j < 8; ++j) { af[i][j] = lo[j]; af[i][j + 8] = hi[j]; }
    }
#pragma unroll
    for (int i = 0; i < WN; ++i) {
      int r = wn * (BN / 4) + i * 16 + lrow;
      v8h lo = *(const v8h*)(Bs + r * 32 + koff);
      v8h hi = *(const v8h*)(Bs + r * 32 + koff + 16);
#pragma unroll
      for (int j = 0; j < 8; ++j) { bf[i][j] = lo[j]; bf[i][j + 8] = hi[j]; }
    }
#pragma unroll
    for (int i = 0; i < WM; ++i)
#pragma unroll
      for (int j = 0; j < WN; ++j)
        acc[i][j] = __builtin_amdgcn_wmma_f32_16x16x32_f16(
            false, af[i], false, bf[j], (short)0, acc[i][j], false, false);
    __syncthreads();
  }

#pragma unroll
  for (int i = 0; i < WM; ++i)
#pragma unroll
    for (int j = 0; j < WN; ++j) {
      int col = col0 + wn * (BN / 4) + j * 16 + lrow;
      int rb = row0 + wm * (BM / 2) + i * 16 + ((lane < 16) ? 0 : 8);
      float bv = ob[col];
#pragma unroll
      for (int r = 0; r < 8; ++r)
        R[(long long)(rb + r) * C_ + col] = (_Float16)(acc[i][j][r] + bv);
    }
}

// ---------------------------------------------------------------------------
// Scatter qkv (B*D*T, 3*C) into head layouts (q pre-scaled; vt transposed F x T)
// ---------------------------------------------------------------------------
__global__ void k_qkv_scatter(const float* __restrict__ qkv,
                              _Float16* __restrict__ qh, _Float16* __restrict__ kh,
                              _Float16* __restrict__ ksh, _Float16* __restrict__ vt) {
  long long i = (long long)blockIdx.x * 256 + threadIdx.x;
  if (i >= (long long)B_ * D_ * T_ * 3 * C_) return;
  int n = (int)(i % (3 * C_));
  long long m = i / (3 * C_);
  int t = (int)(m % T_);
  int bd = (int)(m / T_);
  int which = n / C_, h = (n % C_) / F_, f = n % F_;
  float v = qkv[i];
  long long o = (((long long)bd * H_ + h) * T_ + t) * F_ + f;
  if (which == 0) qh[o] = (_Float16)(v * SCALE_);
  else if (which == 1) { kh[o] = (_Float16)v; ksh[o] = (_Float16)(v * SCALE_); }
  else vt[(((long long)bd * H_ + h) * F_ + f) * T_ + t] = (_Float16)v;
}

// ---------------------------------------------------------------------------
// attn[b,d,h,t,s] += sum_f Q[..]*R[..]   mode 0: q.Rk   mode 1: (k*scale).Rq
// ---------------------------------------------------------------------------
__global__ void __launch_bounds__(192) k_add_qR(float* __restrict__ attn,
                                                const _Float16* __restrict__ Q,
                                                const _Float16* __restrict__ R,
                                                int mode) {
  int bi = blockIdx.x;
  int t = bi % T_; bi /= T_;
  int h = bi % H_; bi /= H_;
  int d = bi % D_;
  int b = bi / D_;
  int s = threadIdx.x;
  const _Float16* qp;
  const _Float16* rp;
  if (mode == 0) {
    qp = Q + ((((long long)(b * D_ + d) * H_ + h) * T_ + t) * F_);
    rp = R + (((long long)(b * T_ + t) * T_ + s) * C_ + h * F_);
  } else {
    qp = Q + ((((long long)(b * D_ + d) * H_ + h) * T_ + s) * F_);
    rp = R + (((long long)(b * T_ + s) * T_ + t) * C_ + h * F_);
  }
  float acc = 0.f;
#pragma unroll
  for (int f = 0; f < F_; f += 8) {
    v8h qv = *(const v8h*)(qp + f);
    v8h rv = *(const v8h*)(rp + f);
#pragma unroll
    for (int j = 0; j < 8; ++j) acc += (float)qv[j] * (float)rv[j];
  }
  attn[((((long long)(b * D_ + d) * H_ + h) * T_ + t) * T_) + s] += acc;
}

// ---------------------------------------------------------------------------
// Masked softmax over s, writes f16 probabilities
// ---------------------------------------------------------------------------
__global__ void __launch_bounds__(256) k_softmax(const float* __restrict__ attn,
                                                 _Float16* __restrict__ attnf,
                                                 const int* __restrict__ mask) {
  __shared__ float red[256];
  int bi = blockIdx.x;
  int t = bi % T_;
  int h = (bi / T_) % H_;
  int d = (bi / (T_ * H_)) % D_;
  int b = bi / (T_ * H_ * D_);
  long long base = ((((long long)(b * D_ + d) * H_ + h) * T_ + t) * T_);
  int s = threadIdx.x;
  int mt = mask[b * T_ + t];
  bool ok = false;
  float v = -3.0e38f;
  if (s < T_) {
    ok = (mask[b * T_ + s] == mt);
    v = ok ? attn[base + s] : -3.0e38f;
  }
  red[s] = v;
  __syncthreads();
  for (int o = 128; o > 0; o >>= 1) {
    if (s < o) red[s] = fmaxf(red[s], red[s + o]);
    __syncthreads();
  }
  float mx = red[0];
  __syncthreads();
  float e = (s < T_ && ok) ? __expf(v - mx) : 0.f;
  red[s] = e;
  __syncthreads();
  for (int o = 128; o > 0; o >>= 1) {
    if (s < o) red[s] += red[s + o];
    __syncthreads();
  }
  float inv = 1.f / red[0];
  if (s < T_) attnf[base + s] = (_Float16)(e * inv);
}

// ---------------------------------------------------------------------------
// outh[b,d,h,t,f] += sum_s attnf[b,d,h,t,s] * Rv[b,t,s,h,f]
// ---------------------------------------------------------------------------
__global__ void __launch_bounds__(64) k_add_Rv(float* __restrict__ outh,
                                               const _Float16* __restrict__ attnf,
                                               const _Float16* __restrict__ R) {
  __shared__ float sa[T_];
  int bi = blockIdx.x;
  int t = bi % T_;
  int h = (bi / T_) % H_;
  int d = (bi / (T_ * H_)) % D_;
  int b = bi / (T_ * H_ * D_);
  long long abase = ((((long long)(b * D_ + d) * H_ + h) * T_ + t) * T_);
  int f = threadIdx.x;
  for (int i = f; i < T_; i += 64) sa[i] = (float)attnf[abase + i];
  __syncthreads();
  const _Float16* rp = R + (((long long)(b * T_ + t) * T_) * C_ + h * F_ + f);
  float acc = 0.f;
  for (int s = 0; s < T_; ++s) acc += sa[s] * (float)rp[(long long)s * C_];
  long long o = ((((long long)(b * D_ + d) * H_ + h) * T_ + t) * F_) + f;
  outh[o] += acc;
}

// ---------------------------------------------------------------------------
// (B,D,H,T,F) f32 -> (B,D,T,C) f16
// ---------------------------------------------------------------------------
__global__ void k_merge(const float* __restrict__ outh, _Float16* __restrict__ outa) {
  long long i = (long long)blockIdx.x * 256 + threadIdx.x;
  if (i >= (long long)B_ * D_ * H_ * T_ * F_) return;
  int f = (int)(i % F_);
  long long r = i / F_;
  int t = (int)(r % T_); r /= T_;
  int h = (int)(r % H_);
  int bd = (int)(r / H_);
  outa[(((long long)bd * T_ + t) * C_) + h * F_ + f] = (_Float16)outh[i];
}

// ---------------------------------------------------------------------------
// out(B,D,C,T) = transpose(xt + proj)
// ---------------------------------------------------------------------------
__global__ void k_final(const float* __restrict__ xt, const float* __restrict__ proj,
                        float* __restrict__ out) {
  long long i = (long long)blockIdx.x * 256 + threadIdx.x;
  if (i >= (long long)B_ * D_ * T_ * C_) return;
  int c = (int)(i % C_);
  long long r = i / C_;
  int t = (int)(r % T_);
  int bd = (int)(r / T_);
  out[(((long long)bd * C_ + c) * T_) + t] = xt[i] + proj[i];
}

// ---------------------------------------------------------------------------
extern "C" void kernel_launch(void* const* d_in, const int* in_sizes, int n_in,
                              void* d_out, int out_size, void* d_ws, size_t ws_size,
                              hipStream_t stream) {
  (void)in_sizes; (void)n_in; (void)out_size; (void)ws_size;
  const float* x      = (const float*)d_in[0];
  const float* femb   = (const float*)d_in[1];
  const int*   fdi    = (const int*)d_in[2];
  const int*   maskp  = (const int*)d_in[3];
  const float* gn_w   = (const float*)d_in[4];
  const float* gn_b   = (const float*)d_in[5];
  const float* qkv_w  = (const float*)d_in[6];
  const float* qkv_b  = (const float*)d_in[7];
  const float* proj_w = (const float*)d_in[8];
  const float* proj_b = (const float*)d_in[9];
  const float* dist_w[3] = {(const float*)d_in[10], (const float*)d_in[16], (const float*)d_in[22]};
  const float* dist_b[3] = {(const float*)d_in[11], (const float*)d_in[17], (const float*)d_in[23]};
  const float* time_w[3] = {(const float*)d_in[12], (const float*)d_in[18], (const float*)d_in[24]};
  const float* time_b[3] = {(const float*)d_in[13], (const float*)d_in[19], (const float*)d_in[25]};
  const float* out_w[3]  = {(const float*)d_in[14], (const float*)d_in[20], (const float*)d_in[26]};
  const float* out_b[3]  = {(const float*)d_in[15], (const float*)d_in[21], (const float*)d_in[27]};
  float* out = (float*)d_out;

  size_t off = 0;
  auto alloc = [&](size_t bytes) -> char* {
    char* p = (char*)d_ws + off;
    off += (bytes + 255) & ~(size_t)255;
    return p;
  };

  const long long NE = (long long)B_ * D_ * T_ * C_;             // 786432
  const long long NA = (long long)B_ * D_ * H_ * T_ * T_;        // 2359296
  const long long NR = (long long)B_ * T_ * T_ * C_;             // 37748736

  _Float16* xth     = (_Float16*)alloc(NE * 2);
  float*    xt      = (float*)alloc(NE * 4);
  _Float16* qkv_wt  = (_Float16*)alloc((size_t)1536 * 512 * 2);
  _Float16* proj_wt = (_Float16*)alloc((size_t)512 * 512 * 2);
  _Float16* time_wt[3], *out_wt[3];
  for (int p = 0; p < 3; ++p) time_wt[p] = (_Float16*)alloc((size_t)512 * 512 * 2);
  for (int p = 0; p < 3; ++p) out_wt[p]  = (_Float16*)alloc((size_t)512 * 512 * 2);
  _Float16* fembh   = (_Float16*)alloc((size_t)B_ * T_ * TE_ * 2);
  float*    qkvbuf  = (float*)alloc((size_t)1536 * 1536 * 4);
  _Float16* qh      = (_Float16*)alloc(NE * 2);
  _Float16* kh      = (_Float16*)alloc(NE * 2);
  _Float16* ksh     = (_Float16*)alloc(NE * 2);
  _Float16* vt      = (_Float16*)alloc(NE * 2);
  float*    temb[3];
  for (int p = 0; p < 3; ++p) temb[p] = (float*)alloc((size_t)B_ * T_ * C_ * 4);
  _Float16* R       = (_Float16*)alloc((size_t)NR * 2);
  float*    attn    = (float*)alloc((size_t)NA * 4);
  _Float16* attnf   = (_Float16*)alloc((size_t)NA * 2);
  float*    outh    = (float*)alloc(NE * 4);
  _Float16* outa    = (_Float16*)alloc(NE * 2);
  float*    projo   = (float*)alloc(NE * 4);

  // ---- weight conversions (f16, transposed to N x K) ----
  k_cvt_t<<<(512 * 1536 + 255) / 256, 256, 0, stream>>>(qkv_w, qkv_wt, 512, 1536);
  k_cvt_t<<<(512 * 512 + 255) / 256, 256, 0, stream>>>(proj_w, proj_wt, 512, 512);
  for (int p = 0; p < 3; ++p) {
    k_cvt_t<<<(512 * 512 + 255) / 256, 256, 0, stream>>>(time_w[p], time_wt[p], 512, 512);
    k_cvt_t<<<(512 * 512 + 255) / 256, 256, 0, stream>>>(out_w[p], out_wt[p], 512, 512);
  }
  k_cvt<<<((int)(B_ * T_ * TE_) + 255) / 256, 256, 0, stream>>>(femb, fembh, B_ * T_ * TE_);

  // ---- GroupNorm + transpose ----
  k_groupnorm<<<B_ * D_ * NG_, 256, 0, stream>>>(x, gn_w, gn_b, xt, xth);

  // ---- QKV GEMM: (1536 x 512) x (512 x 1536) ----
  k_gemm<128, 128, 5><<<dim3(12, 12, 1), 256, 0, stream>>>(
      xth, qkv_wt, qkvbuf, nullptr, qkv_b, 1536, 1536, 512, 0, 0, 0);
  k_qkv_scatter<<<(int)((NA + 255) / 256), 256, 0, stream>>>(qkvbuf, qh, kh, ksh, vt);

  // ---- time-embedding GEMMs: (384 x 512) x (512 x 512) ----
  for (int p = 0; p < 3; ++p)
    k_gemm<128, 128, 5><<<dim3(4, 3, 1), 256, 0, stream>>>(
        fembh, time_wt[p], temb[p], nullptr, time_b[p], B_ * T_, C_, TE_, 0, 0, 0);

  // ---- attn = q k^T, batched over B*D*H = 64 ----
  k_gemm<64, 64, 4><<<dim3(3, 3, B_ * D_ * H_), 256, 0, stream>>>(
      qh, kh, attn, nullptr, nullptr, T_, T_, F_,
      (long long)T_ * F_, (long long)T_ * F_, (long long)T_ * T_);

  const int nRow = B_ * D_ * H_ * T_;  // 12288

  // ---- Rk (p index 1 = k) ----
  k_rpe_gemm<<<dim3(4, 576), 256, 0, stream>>>(temb[1], dist_w[1], dist_b[1], fdi,
                                               out_wt[1], out_b[1], R);
  k_add_qR<<<nRow, 192, 0, stream>>>(attn, qh, R, 0);

  // ---- Rq (p index 0 = q) ----
  k_rpe_gemm<<<dim3(4, 576), 256, 0, stream>>>(temb[0], dist_w[0], dist_b[0], fdi,
                                               out_wt[0], out_b[0], R);
  k_add_qR<<<nRow, 192, 0, stream>>>(attn, ksh, R, 1);

  // ---- softmax ----
  k_softmax<<<nRow, 256, 0, stream>>>(attn, attnf, maskp);

  // ---- out = attn v, batched ----
  k_gemm<64, 64, 4><<<dim3(1, 3, B_ * D_ * H_), 256, 0, stream>>>(
      attnf, vt, outh, nullptr, nullptr, T_, F_, T_,
      (long long)T_ * T_, (long long)F_ * T_, (long long)T_ * F_);

  // ---- Rv (p index 2 = v) ----
  k_rpe_gemm<<<dim3(4, 576), 256, 0, stream>>>(temb[2], dist_w[2], dist_b[2], fdi,
                                               out_wt[2], out_b[2], R);
  k_add_Rv<<<nRow, 64, 0, stream>>>(outh, attnf, R);

  // ---- merge heads + proj + residual/transpose ----
  k_merge<<<(int)((NE + 255) / 256), 256, 0, stream>>>(outh, outa);
  k_gemm<128, 128, 5><<<dim3(4, 12, 1), 256, 0, stream>>>(
      outa, proj_wt, projo, nullptr, proj_b, B_ * D_ * T_, C_, C_, 0, 0, 0);
  k_final<<<(int)((NE + 255) / 256), 256, 0, stream>>>(xt, projo, out);
}
